// BinaryLinear_35691178230429
// MI455X (gfx1250) — matrix operand order repaired and checked by reading, not compiled, not run on hardware
//
#include <hip/hip_runtime.h>
#include <hip/hip_bf16.h>

// ---------------------------------------------------------------------------
// BinaryLinear on gfx1250:  out = x @ (scale * sign(W))^T + bias
//   x: [16384, 4096] f32   W: [4096, 4096] f32   bias: [4096] f32
// v3: async global->LDS staging (global_load_async_to_lds_b128 + s_wait_asynccnt),
//     double-buffered LDS, 64x64 per-wave tiles on v_wmma_f32_16x16x32_f16,
//     non-temporal stores for the streamed output (preserve L2 for Xb/Wb reuse).
// ---------------------------------------------------------------------------

#define IN_F   4096
#define OUT_F  4096
#define M_TOT  16384

typedef __attribute__((ext_vector_type(16))) _Float16 v16h;
typedef __attribute__((ext_vector_type(8)))  _Float16 v8h;
typedef __attribute__((ext_vector_type(8)))  float    v8f;
typedef __attribute__((ext_vector_type(4)))  float    v4f;

// ---------------------------------------------------------------------------
// Kernel 1: per-row |w| mean + sign binarization to f16.  One wave per row.
// w is read exactly once -> non-temporal loads.
// ---------------------------------------------------------------------------
__global__ __launch_bounds__(256)
void binarize_weight_kernel(const float* __restrict__ w,
                            _Float16* __restrict__ wb,
                            float* __restrict__ scale) {
    const int wid  = threadIdx.x >> 5;
    const int lane = threadIdx.x & 31;
    const int row  = blockIdx.x * 8 + wid;

    const float* wr  = w  + (size_t)row * IN_F;
    _Float16*    wbr = wb + (size_t)row * IN_F;

    float s = 0.0f;
    for (int k = lane; k < IN_F; k += 32) {
        const float v = __builtin_nontemporal_load(wr + k);
        s += fabsf(v);
        const float sg = (v > 0.0f) ? 1.0f : ((v < 0.0f) ? -1.0f : 0.0f);
        wbr[k] = (_Float16)sg;          // Wb is re-read by the GEMM: keep RT
    }
    #pragma unroll
    for (int o = 16; o > 0; o >>= 1) s += __shfl_xor(s, o, 32);
    if (lane == 0) scale[row] = s * (1.0f / (float)IN_F);
}

// ---------------------------------------------------------------------------
// Kernel 2: x fp32 -> f16 (RNE), 8 elements / thread.
// x fp32 is read exactly once -> non-temporal; Xb f16 is re-read -> RT store.
// ---------------------------------------------------------------------------
__global__ __launch_bounds__(256)
void convert_x_kernel(const float* __restrict__ x, _Float16* __restrict__ xb) {
    const size_t i = ((size_t)blockIdx.x * blockDim.x + threadIdx.x) * 8;
    const v4f a = __builtin_nontemporal_load((const v4f*)(x + i));
    const v4f b = __builtin_nontemporal_load((const v4f*)(x + i + 4));
    v8h o;
    o[0] = (_Float16)a[0]; o[1] = (_Float16)a[1];
    o[2] = (_Float16)a[2]; o[3] = (_Float16)a[3];
    o[4] = (_Float16)b[0]; o[5] = (_Float16)b[1];
    o[6] = (_Float16)b[2]; o[7] = (_Float16)b[3];
    *(v8h*)(xb + i) = o;
}

// ---------------------------------------------------------------------------
// Kernel 3: WMMA GEMM with async double-buffered staging.
//   Block tile 128(M) x 256(N), BK=32.  256 threads = 8 waves (2x4).
//   Each wave: 64x64 sub-tile = 4x4 f32 accumulator fragments.
//   LDS rows padded to 80 B: 16B-aligned and 20-bank step -> the 16 lanes of a
//   fragment ds_load_b128 hit 16 distinct bank groups (conflict-free).
// ---------------------------------------------------------------------------
#define BM  128
#define BN  256
#define BK  32
#define LDT 40                    // halves per LDS tile row (32 data + 8 pad)
#define AS_HALF (BM * LDT)        // halves per X buffer
#define BS_HALF (BN * LDT)        // halves per W buffer

// async copy: 16 B per lane, global (SGPR base + 32-bit lane offset) -> LDS
#define ASYNC_LD128(ldsaddr, goff, base)                                   \
    asm volatile("global_load_async_to_lds_b128 %0, %1, %2"                \
                 :: "v"(ldsaddr), "v"(goff), "s"(base) : "memory")

#define WAIT_ASYNCCNT(n)                                                   \
    asm volatile("s_wait_asynccnt %0" :: "n"(n) : "memory")

__global__ __launch_bounds__(256)
void bitlinear_gemm_kernel(const _Float16* __restrict__ Xb,
                           const _Float16* __restrict__ Wb,
                           const float* __restrict__ scale,
                           const float* __restrict__ bias,
                           float* __restrict__ out) {
    __shared__ alignas(16) _Float16 As[2 * AS_HALF];
    __shared__ alignas(16) _Float16 Bs[2 * BS_HALF];

    const int tid    = threadIdx.x;
    const int wid    = tid >> 5;
    const int lane   = tid & 31;
    const int wave_m = wid & 1;     // 2 slabs of 64 rows
    const int wave_n = wid >> 1;    // 4 slabs of 64 cols
    const int m0 = blockIdx.y * BM;
    const int n0 = blockIdx.x * BN;

    v8f acc[4][4] = {};
    union Frag { v16h v; uint4 q[2]; };

    const int lhi   = lane >> 4;
    const int l16   = lane & 15;
    const int ahalf = lhi * 8;      // A frag K chunks {0-7,16-23} / {8-15,24-31}
    const int bhalf = lhi * 8;      // B frag K chunks {0-7,16-23} / {8-15,24-31}

    // ---- per-thread staging descriptors: 6 x 16B chunks per K-step ----
    const int      crow = tid >> 2;         // 0..63
    const int      cseg = (tid & 3) * 8;    // halves
    const unsigned asb  = (unsigned)(uintptr_t)As;
    const unsigned bsb  = (unsigned)(uintptr_t)Bs;

    const unsigned ldsX0 = asb + (unsigned)((crow        * LDT + cseg) * 2);
    const unsigned ldsX1 = asb + (unsigned)(((crow + 64) * LDT + cseg) * 2);
    const unsigned ldsW0 = bsb + (unsigned)((crow         * LDT + cseg) * 2);
    const unsigned ldsW1 = bsb + (unsigned)(((crow +  64) * LDT + cseg) * 2);
    const unsigned ldsW2 = bsb + (unsigned)(((crow + 128) * LDT + cseg) * 2);
    const unsigned ldsW3 = bsb + (unsigned)(((crow + 192) * LDT + cseg) * 2);

    const unsigned gX0 = (unsigned)(((unsigned)(m0 + crow) * IN_F + cseg) * 2);
    const unsigned gX1 = gX0 + 64u * IN_F * 2u;
    const unsigned gW0 = (unsigned)(((unsigned)(n0 + crow) * IN_F + cseg) * 2);
    const unsigned gW1 = gW0 +  64u * IN_F * 2u;
    const unsigned gW2 = gW0 + 128u * IN_F * 2u;
    const unsigned gW3 = gW0 + 192u * IN_F * 2u;

    auto issue = [&](int buf, unsigned kb) {
        const unsigned ao = buf ? (unsigned)(AS_HALF * 2) : 0u;
        const unsigned bo = buf ? (unsigned)(BS_HALF * 2) : 0u;
        ASYNC_LD128(ldsX0 + ao, gX0 + kb, Xb);
        ASYNC_LD128(ldsX1 + ao, gX1 + kb, Xb);
        ASYNC_LD128(ldsW0 + bo, gW0 + kb, Wb);
        ASYNC_LD128(ldsW1 + bo, gW1 + kb, Wb);
        ASYNC_LD128(ldsW2 + bo, gW2 + kb, Wb);
        ASYNC_LD128(ldsW3 + bo, gW3 + kb, Wb);
    };

    auto compute = [&](int buf) {
        const _Float16* asp = As + buf * AS_HALF;
        const _Float16* bsp = Bs + buf * BS_HALF;
        Frag a[4];
        #pragma unroll
        for (int i = 0; i < 4; ++i) {
            const _Float16* p = asp + (wave_m * 64 + i * 16 + l16) * LDT;
            a[i].q[0] = *(const uint4*)(p + ahalf);
            a[i].q[1] = *(const uint4*)(p + 16 + ahalf);
        }
        #pragma unroll
        for (int j = 0; j < 4; ++j) {
            const _Float16* p = bsp + (wave_n * 64 + j * 16 + l16) * LDT + bhalf;
            Frag b;
            b.q[0] = *(const uint4*)(p);
            b.q[1] = *(const uint4*)(p + 16);
            #pragma unroll
            for (int i = 0; i < 4; ++i)
                acc[i][j] = __builtin_amdgcn_wmma_f32_16x16x32_f16(
                    false, a[i].v, false, b.v, (short)0, acc[i][j], false, false);
        }
    };

    // ---- software pipeline: issue(k+1) overlaps compute(k) ----
    issue(0, 0u);
    for (int kt = 0; kt < (IN_F / BK) - 1; ++kt) {
        issue((kt + 1) & 1, (unsigned)(kt + 1) * (BK * 2));
        WAIT_ASYNCCNT(6);     // in-order: the 6 older asyncs (tile kt) are done
        __syncthreads();      // all waves' tile-kt data visible in LDS
        compute(kt & 1);
        __syncthreads();      // done reading buf before iter kt+1 overwrites it
    }
    WAIT_ASYNCCNT(0);
    __syncthreads();
    compute(((IN_F / BK) - 1) & 1);

    // ---- epilogue: out = scale[n]*acc + bias[n] (fp32), streamed (TH=NT) ----
    #pragma unroll
    for (int j = 0; j < 4; ++j) {
        const int n = n0 + wave_n * 64 + j * 16 + l16;
        const float sc = scale[n];
        const float bi = bias[n];
        #pragma unroll
        for (int i = 0; i < 4; ++i) {
            const int mb = m0 + wave_m * 64 + i * 16 + lhi * 8;
            #pragma unroll
            for (int r = 0; r < 8; ++r)
                __builtin_nontemporal_store(acc[i][j][r] * sc + bi,
                                            out + (size_t)(mb + r) * OUT_F + n);
        }
    }
}

// ---------------------------------------------------------------------------
extern "C" void kernel_launch(void* const* d_in, const int* in_sizes, int n_in,
                              void* d_out, int out_size, void* d_ws, size_t ws_size,
                              hipStream_t stream) {
    (void)in_sizes; (void)n_in; (void)out_size; (void)ws_size;
    const float* x      = (const float*)d_in[0];
    const float* weight = (const float*)d_in[1];
    const float* bias   = (const float*)d_in[2];
    float* out = (float*)d_out;

    char* ws = (char*)d_ws;
    float*    scale = (float*)ws;                                   // 16 KB
    _Float16* Wb    = (_Float16*)(ws + 16384);                      // 32 MB
    _Float16* Xb    = (_Float16*)(ws + 16384 + (size_t)OUT_F * IN_F * sizeof(_Float16));

    binarize_weight_kernel<<<OUT_F / 8, 256, 0, stream>>>(weight, Wb, scale);

    const int conv_blocks = (int)(((size_t)M_TOT * IN_F) / (8 * 256));  // 32768
    convert_x_kernel<<<conv_blocks, 256, 0, stream>>>(x, Xb);

    dim3 grid(OUT_F / BN, M_TOT / BM);  // (16, 128)
    bitlinear_gemm_kernel<<<grid, 256, 0, stream>>>(Xb, Wb, scale, bias, out);
}
